// ChordProgressionLoss_9766755631310
// MI455X (gfx1250) — compile-verified, matrix operand order checked by reading, not run on hardware
//
#include <hip/hip_runtime.h>

typedef __attribute__((ext_vector_type(16))) _Float16     v16h;
typedef __attribute__((ext_vector_type(8)))  float        v8f;
typedef __attribute__((ext_vector_type(8)))  unsigned int v8u;

#define BLOCK_ROWS 256

#if defined(__HIP_DEVICE_COMPILE__) && __has_builtin(__builtin_amdgcn_wmma_f32_16x16x32_f16)
#  define CHORD_USE_WMMA 1
#else
#  define CHORD_USE_WMMA 0
#endif

// Pitch-class set of 4 notes as a 12-bit mask (bits >= 12 never set).
__device__ __forceinline__ unsigned pc_mask4(float4 v) {
    unsigned m = 0u;
    m |= 1u << (((unsigned)(int)v.x) % 12u);
    m |= 1u << (((unsigned)(int)v.y) % 12u);
    m |= 1u << (((unsigned)(int)v.z) % 12u);
    m |= 1u << (((unsigned)(int)v.w) % 12u);
    return m;
}

// 8 template masks (n=0..3 major j, n=4..7 minor j), packed in two u64 constants.
// major: {0,4,7}=0x091 {5,9,0}=0x221 {7,11,2}=0x884 {0,4,7}=0x091
// minor: {0,3,7}=0x089 {5,8,0}=0x121 {7,10,2}=0x484 {0,3,7}=0x089
__device__ __forceinline__ unsigned templ_mask(int n) {
    const unsigned long long lo = 0x0091088402210091ULL;
    const unsigned long long hi = 0x0089048401210089ULL;
    unsigned long long w = (n < 4) ? lo : hi;
    return (unsigned)((w >> (16 * (n & 3))) & 0xFFFu);
}

// Two mask bits (k0, k0+1) -> packed pair of f16 {0.0, 1.0} in a 32-bit word.
__device__ __forceinline__ unsigned bits_to_h2(unsigned mask, int k0) {
    unsigned lo = ((mask >> k0) & 1u) ? 0x3C00u : 0u;
    unsigned hi = ((mask >> (k0 + 1)) & 1u) ? 0x3C00u : 0u;
    return lo | (hi << 16);
}

__device__ __forceinline__ float fast_rcp(float x) {
    return __builtin_amdgcn_rcpf(x);
}

__global__ void __launch_bounds__(256) chord_loss_main(
    const float* __restrict__ pred, const float* __restrict__ targ,
    float* __restrict__ partial, int T)
{
    __shared__ float dtab[BLOCK_ROWS + 3][8];  // Jaccard distances vs 8 templates
    __shared__ float red[256];

    const int tid  = threadIdx.x;
    const int lane = tid & 31;
    const int wave = tid >> 5;
    const int blockStart = blockIdx.x * BLOCK_ROWS;
    const int row = blockStart + tid;

    unsigned P = 0u, Q = 0u;
    int szP = 0, szQ = 0;
    if (row < T) {
        float4 pv = ((const float4*)pred)[row];
        float4 tv = ((const float4*)targ)[row];
        P = pc_mask4(pv); Q = pc_mask4(tv);
        szP = __popc(P); szQ = __popc(Q);
    }

    // ---- similarity term (per row, popcount exact; rcp instead of div) ----
    float simv = 0.0f;
    if (row < T) {
        const float eps = 1e-6f;
        float inter = (float)__popc(P & Q);
        float num = inter + eps * (float)(szP + szQ) + 12.0f * eps * eps;
        float den = ((float)szP + 12.0f * eps) * ((float)szQ + 12.0f * eps);
        float s = num * fast_rcp(den);
        simv = fminf(fmaxf(s, 0.0f), 1.0f);
    }

    const int hi = lane >> 4;   // which 16-lane half
    const int n  = lane & 15;   // template/column index in WMMA result

#if CHORD_USE_WMMA
    // B operand (32x16 f16): templates on N=0..7, zero elsewhere.
    // Layout: element 2v+j of lane l carries K = (v&4?16:0) + 2*(v&3) + 8*(l>=16) + j.
    // Template masks have no bits >= 12, so padded-K words are naturally zero.
    v16h b;
    {
        unsigned tm = (n < 8) ? templ_mask(n) : 0u;
        v8u bw;
        #pragma unroll
        for (int v = 0; v < 8; ++v) {
            int k0 = ((v & 4) ? 16 : 0) + 2 * (v & 3) + 8 * hi;
            bw[v] = bits_to_h2(tm, k0);
        }
        b = __builtin_bit_cast(v16h, bw);
    }

    // Two 16-row groups per wave: rows blockStart + wave*32 + 16*g + m.
    #pragma unroll
    for (int g = 0; g < 2; ++g) {
        // lane l needs the mask of row m = l&15 of this group, owned by lane 16*g + (l&15)
        unsigned Pm = (unsigned)__shfl((int)P, 16 * g + (lane & 15), 32);
        v16h a;
        {
            v8u aw;
            #pragma unroll
            for (int v = 0; v < 8; ++v) {
                int k0 = ((v & 4) ? 16 : 0) + 2 * (v & 3) + 8 * hi;
                aw[v] = bits_to_h2(Pm, k0);
            }
            a = __builtin_bit_cast(v16h, aw);
        }
        v8f cacc = {};
        cacc = __builtin_amdgcn_wmma_f32_16x16x32_f16(
            /*neg_a=*/false, a, /*neg_b=*/false, b,
            /*c_mod=*/(short)0, cacc, /*reuse_a=*/false, /*reuse_b=*/false);

        // D: VGPR r, lane l -> inter[m = r + 8*hi][n = l&15]; exact small ints.
        float dvals[8];
        #pragma unroll
        for (int r = 0; r < 8; ++r) {
            int m = r + 8 * hi;
            int szm = __shfl(szP, 16 * g + m, 32);   // all lanes active here
            float inter = cacc[r];
            float uni = (float)szm + 3.0f - inter + 1e-8f;
            dvals[r] = 1.0f - inter * fast_rcp(uni);
        }
        if (n < 8) {
            #pragma unroll
            for (int r = 0; r < 8; ++r)
                dtab[wave * 32 + 16 * g + (r + 8 * hi)][n] = dvals[r];
        }
    }
#else
    #pragma unroll
    for (int nn = 0; nn < 8; ++nn) {
        float inter = (float)__popc(P & templ_mask(nn));
        float uni = (float)szP + 3.0f - inter + 1e-8f;
        dtab[tid][nn] = 1.0f - inter * fast_rcp(uni);
    }
#endif

    // ---- halo rows (3 per block) via popcount (numerically identical) ----
    if (tid < 3) {
        int hrow = blockStart + BLOCK_ROWS + tid;
        unsigned Ph = 0u; int szh = 0;
        if (hrow < T) {
            float4 pv = ((const float4*)pred)[hrow];
            Ph = pc_mask4(pv); szh = __popc(Ph);
        }
        #pragma unroll
        for (int nn = 0; nn < 8; ++nn) {
            float inter = (float)__popc(Ph & templ_mask(nn));
            float uni = (float)szh + 3.0f - inter + 1e-8f;
            dtab[BLOCK_ROWS + tid][nn] = 1.0f - inter * fast_rcp(uni);
        }
    }
    __syncthreads();

    // ---- sliding 4-window progression penalty ----
    float pen = 0.0f;
    if (row <= T - 4) {
        float maj = 0.25f * (dtab[tid][0] + dtab[tid + 1][1] + dtab[tid + 2][2] + dtab[tid + 3][3]);
        float mio = 0.25f * (dtab[tid][4] + dtab[tid + 1][5] + dtab[tid + 2][6] + dtab[tid + 3][7]);
        pen = fminf(maj, mio);
    }

    // ---- deterministic block reduction of (sim, pen) ----
    red[tid] = simv;
    __syncthreads();
    for (int s = 128; s > 0; s >>= 1) {
        if (tid < s) red[tid] += red[tid + s];
        __syncthreads();
    }
    if (tid == 0) partial[2 * blockIdx.x + 0] = red[0];
    __syncthreads();
    red[tid] = pen;
    __syncthreads();
    for (int s = 128; s > 0; s >>= 1) {
        if (tid < s) red[tid] += red[tid + s];
        __syncthreads();
    }
    if (tid == 0) partial[2 * blockIdx.x + 1] = red[0];
}

__global__ void __launch_bounds__(256) chord_loss_final(
    const float* __restrict__ partial, float* __restrict__ out, int nBlocks, int T)
{
    __shared__ float rs[256], rp[256];
    float s = 0.0f, p = 0.0f;
    for (int i = threadIdx.x; i < nBlocks; i += 256) {
        s += partial[2 * i + 0];
        p += partial[2 * i + 1];
    }
    rs[threadIdx.x] = s; rp[threadIdx.x] = p;
    __syncthreads();
    for (int st = 128; st > 0; st >>= 1) {
        if (threadIdx.x < st) {
            rs[threadIdx.x] += rs[threadIdx.x + st];
            rp[threadIdx.x] += rp[threadIdx.x + st];
        }
        __syncthreads();
    }
    if (threadIdx.x == 0) {
        float simLoss = 1.0f - rs[0] / (float)T;
        float penalty = rp[0] / (float)(T - 3);
        out[0] = simLoss + 0.5f * penalty;
    }
}

extern "C" void kernel_launch(void* const* d_in, const int* in_sizes, int n_in,
                              void* d_out, int out_size, void* d_ws, size_t ws_size,
                              hipStream_t stream) {
    (void)n_in; (void)out_size; (void)ws_size;
    const float* pred = (const float*)d_in[0];
    const float* targ = (const float*)d_in[1];
    float* out = (float*)d_out;
    float* ws  = (float*)d_ws;   // 2 floats per block of partial sums (~31 KB)
    const int T = in_sizes[0] / 4;
    const int nBlocks = (T + BLOCK_ROWS - 1) / BLOCK_ROWS;
    chord_loss_main<<<nBlocks, 256, 0, stream>>>(pred, targ, ws, T);
    chord_loss_final<<<1, 256, 0, stream>>>(ws, out, nBlocks, T);
}
